// pointconv_old_81501299409046
// MI455X (gfx1250) — matrix operand order repaired and checked by reading, not compiled, not run on hardware
//
#include <hip/hip_runtime.h>

// ---------------------------------------------------------------------------
// PointConv fused pipeline for MI455X (gfx1250, wave32, WMMA bf16).
//
// Stage-1: per-point [64x32]@[32x32] via v_wmma_f32_16x16x32_bf16 (8/point);
//          D tiles packed to bf16 pairs (permuted K layout) and staged in LDS,
//          then copied out with coalesced b128 stores.
// Stage-2: [16384 x 128] = [16384 x 2048] @ [2048 x 128] bf16 GEMM.
//          8 waves/block = 4 strips x 2 column-halves; B chunk (32x128 bf16)
//          double-buffered in LDS via global_load_async_to_lds_b128.
// K layout: within each c-group of 32, memory position q holds m = (q>>1)+16*(q&1);
//          applied identically to mm and Wb, so the contraction is unchanged.
// Workspace: Wb (bf16, 512 KB) + mm (bf16, 64 MB) => ~64.6 MB of d_ws.
// ---------------------------------------------------------------------------

typedef unsigned short u16;
typedef __attribute__((ext_vector_type(16))) __bf16 v16bf;
typedef __attribute__((ext_vector_type(2)))  __bf16 v2bf;
typedef __attribute__((ext_vector_type(2)))  float  v2f;
typedef __attribute__((ext_vector_type(8)))  float  v8f;
typedef int v4i __attribute__((vector_size(16)));

#define BATCH 16
#define NP    1024
#define NS    32
#define CIN   64
#define COUT  128
#define CMID  32
#define PTS   (BATCH * NP)    // 16384
#define KDIM  (CIN * CMID)    // 2048

// ---- gfx1250 async global->LDS path (probe-verified on this toolchain) ----
#if defined(__HIP_DEVICE_COMPILE__) && defined(__has_builtin)
#if __has_builtin(__builtin_amdgcn_global_load_async_to_lds_b128)
#define HAVE_ASYNC_LDS 1
#endif
#endif
#ifndef HAVE_ASYNC_LDS
#define HAVE_ASYNC_LDS 0
#endif

#if HAVE_ASYNC_LDS
typedef __attribute__((address_space(1))) v4i* gv4p;   // global v4i*
typedef __attribute__((address_space(3))) v4i* lv4p;   // LDS v4i*
__device__ __forceinline__ void async_g2l_b128(const void* g, void* l) {
  __builtin_amdgcn_global_load_async_to_lds_b128(
      (gv4p)(unsigned long long)g,
      (lv4p)(unsigned)(unsigned long long)l, 0, 0);
}
__device__ __forceinline__ void wait_async0() {
#if __has_builtin(__builtin_amdgcn_s_wait_asynccnt)
  __builtin_amdgcn_s_wait_asynccnt(0);
#else
  asm volatile("s_wait_asynccnt 0x0" ::: "memory");
#endif
}
#endif

// float -> bf16 bits, round-to-nearest-even (scalar, used in tiny prep kernel)
__device__ __forceinline__ u16 f2bf(float f) {
  unsigned u = __float_as_uint(f);
  u += 0x7FFFu + ((u >> 16) & 1u);
  return (u16)(u >> 16);
}

// two floats -> packed bf16 pair (lo in [15:0], hi in [31:16]).
// Explicit <2 x float> -> <2 x bfloat> fptrunc: lets the backend select the
// native packed convert (V_CVT_PK_BF16_F32) instead of scalar bit-twiddling.
__device__ __forceinline__ unsigned pack_bf16(float lo, float hi) {
  union { v2bf v; unsigned u; } r;
  v2f t;
  t.x = lo;
  t.y = hi;
  r.v = __builtin_convertvector(t, v2bf);
  return r.u;
}

// Load a 16-element bf16 fragment from two 16B-aligned 8-short groups.
__device__ __forceinline__ v16bf load16bf(const u16* lo, const u16* hi) {
  union { v16bf v; uint4 q[2]; } u;
  u.q[0] = *(const uint4*)lo;
  u.q[1] = *(const uint4*)hi;
  return u.v;
}

// Copy one 32x128 bf16 chunk (8 KB) global -> LDS with 256 threads.
__device__ __forceinline__ void copy_chunk(const u16* __restrict__ src,
                                           u16* dst, int tid) {
#pragma unroll
  for (int t = 0; t < 2; ++t) {
    int e = tid + t * 256;                 // 0..511 x 16B
#if HAVE_ASYNC_LDS
    async_g2l_b128(src + e * 8, dst + e * 8);
#else
    *(uint4*)(dst + e * 8) = *(const uint4*)(src + e * 8);
#endif
  }
}

// ---------------------------------------------------------------------------
// Kernel: output[0] = transpose(new_xyz, (0,2,1))  [B,3,NP] -> [B,NP,3]
// ---------------------------------------------------------------------------
__global__ void xyzT_kernel(const float* __restrict__ new_xyz,
                            float* __restrict__ out0) {
  int idx = blockIdx.x * blockDim.x + threadIdx.x;
  if (idx >= BATCH * NP * 3) return;
  int c = idx % 3;
  int p = (idx / 3) & (NP - 1);
  int b = idx / (3 * NP);
  out0[idx] = new_xyz[(b * 3 + c) * NP + p];
}

// ---------------------------------------------------------------------------
// Kernel: Wb[k'][o] = bf16(out_w[o][k] * bn_scale(o)), permuted K layout.
// ---------------------------------------------------------------------------
__global__ void build_wb_kernel(const float* __restrict__ out_w,
                                const float* __restrict__ out_g,
                                const float* __restrict__ out_v,
                                u16* __restrict__ Wb) {
  int idx = blockIdx.x * blockDim.x + threadIdx.x;   // 0 .. 262143
  int o = idx & (COUT - 1);
  int k = idx >> 7;                   // natural k = c*32 + m
  int m = k & (CMID - 1);
  int kp = (k & ~(CMID - 1)) | (2 * (m & 15) + (m >> 4));   // permuted position
  float s = out_g[o] * rsqrtf(out_v[o] + 1e-5f);
  Wb[(kp >> 5) * (CMID * COUT) + o * CMID + (kp & (CMID - 1))] = f2bf(out_w[(size_t)o * KDIM + k] * s);
}

// ---------------------------------------------------------------------------
// Stage 1: density net + weight net + per-point [64x32]@[32x32] WMMA -> mm
// 8 points / block (1 point per wave), 256 threads.
// ---------------------------------------------------------------------------
__global__ void __launch_bounds__(256) stage1_kernel(
    const float* __restrict__ gxyz,  const float* __restrict__ gfeat,
    const float* __restrict__ wh_w,  const float* __restrict__ wh_b,
    const float* __restrict__ wh_g,  const float* __restrict__ wh_beta,
    const float* __restrict__ wh_m,  const float* __restrict__ wh_v,
    const float* __restrict__ n1_w,  const float* __restrict__ n1_b,
    const float* __restrict__ n1_g,  const float* __restrict__ n1_beta,
    const float* __restrict__ n1_m,  const float* __restrict__ n1_v,
    const float* __restrict__ n2_w,  const float* __restrict__ n2_b,
    const float* __restrict__ n2_g,  const float* __restrict__ n2_beta,
    const float* __restrict__ n2_m,  const float* __restrict__ n2_v,
    u16* __restrict__ mmWS) {
  __shared__ __attribute__((aligned(16))) float scaleS[8][NS];       // 1 KB
  __shared__ __attribute__((aligned(16))) u16   wmatS[8][NS][CMID];  // 16 KB (B-layout: [m][s])
  __shared__ __attribute__((aligned(16))) u16   gfbS[8][CIN][NS];    // 32 KB (A-layout: [c][s]),
                                                                     // reused per-wave as mm staging

  const int tid  = threadIdx.x;
  const int lane = tid & 31;   // sample index s
  const int wave = tid >> 5;   // local point
  const int ptBlock = blockIdx.x * 8;
  const int pt = ptBlock + wave;
  const int bidx = pt >> 10, p = pt & (NP - 1);

  // ---- density -> d2 (sigmoid), lane = s; sample 0 zeroed per reference ----
  const float* gx = gxyz + ((size_t)(bidx * NP + p) * NS + lane) * 3;
  float g0 = 0.f, g1 = 0.f, g2 = 0.f;
  if (lane != 0) { g0 = gx[0]; g1 = gx[1]; g2 = gx[2]; }
  float den = g0 + g1 + g2;
  den = (den < 1e-10f) ? 1e-10f : den;
  float inv = 1.0f / den;
  float mx = inv;
  for (int off = 16; off > 0; off >>= 1)
    mx = fmaxf(mx, __shfl_xor(mx, off, 32));
  float ds = inv / mx;

  float acc2 = 0.f;
  for (int j = 0; j < 16; ++j) {
    float v1 = ds * n1_w[j] + n1_b[j];
    v1 = (v1 - n1_m[j]) * (n1_g[j] * rsqrtf(n1_v[j] + 1e-5f)) + n1_beta[j];
    v1 = fmaxf(v1, 0.f);
    acc2 += v1 * n2_w[j];
  }
  float v2 = acc2 + n2_b[0];
  v2 = (v2 - n2_m[0]) * (n2_g[0] * rsqrtf(n2_v[0] + 1e-5f)) + n2_beta[0];
  scaleS[wave][lane] = 1.0f / (1.0f + __expf(-v2));

  // ---- weight net: w[m][s] stored as [m][s], m-pairs packed ----
  for (int m = 0; m < CMID; m += 2) {
    float w0 = g0 * wh_w[m * 3]     + g1 * wh_w[m * 3 + 1] + g2 * wh_w[m * 3 + 2] + wh_b[m];
    w0 = (w0 - wh_m[m]) * (wh_g[m] * rsqrtf(wh_v[m] + 1e-5f)) + wh_beta[m];
    float w1 = g0 * wh_w[m * 3 + 3] + g1 * wh_w[m * 3 + 4] + g2 * wh_w[m * 3 + 5] + wh_b[m + 1];
    w1 = (w1 - wh_m[m + 1]) * (wh_g[m + 1] * rsqrtf(wh_v[m + 1] + 1e-5f)) + wh_beta[m + 1];
    unsigned wpk = pack_bf16(fmaxf(w0, 0.f), fmaxf(w1, 0.f));
    wmatS[wave][m][lane] = (u16)wpk;
    wmatS[wave][m + 1][lane] = (u16)(wpk >> 16);
  }
  __syncthreads();

  // ---- grouped_feature tile -> LDS: (c, s-pair) tasks, packed b32 stores ----
  const int p0 = ptBlock & (NP - 1), bb = ptBlock >> 10;
  for (int i = 0; i < 4; ++i) {
    int task = tid + i * 256;            // 0..1023 -> (c, sp)
    int c = task >> 4, sp = task & 15;   // covers s = 2sp, 2sp+1
    const float* s0p = gfeat + ((size_t)((bb * CIN + c) * NS + 2 * sp)) * NP + p0;
    const float* s1p = s0p + NP;
    float4 a0 = *(const float4*)s0p;
    float4 a1 = *(const float4*)(s0p + 4);
    float4 b0 = *(const float4*)s1p;
    float4 b1 = *(const float4*)(s1p + 4);
    float va[8] = {a0.x, a0.y, a0.z, a0.w, a1.x, a1.y, a1.z, a1.w};
    float vb[8] = {b0.x, b0.y, b0.z, b0.w, b1.x, b1.y, b1.z, b1.w};
#pragma unroll
    for (int q = 0; q < 8; ++q) {
      unsigned pk = pack_bf16(va[q] * scaleS[q][2 * sp],
                              vb[q] * scaleS[q][2 * sp + 1]);
      *(unsigned*)&gfbS[q][c][2 * sp] = pk;
    }
  }
  __syncthreads();

  // ---- per-point WMMA: mm[64x32] = gfb[64x32] @ wmat[32x32] ----
  // After A rows [ci*16, ci*16+16) are consumed, reuse them to stage D tiles.
  // Permuted K: position pair (2*r16, 2*r16+1) holds (m=r16, m=16+r16).
  const int h = lane >> 4, r16 = lane & 15;
  u16* mmPt  = &gfbS[wave][0][0];              // 64x32 bf16 staging (this wave)
  unsigned* mm32 = (unsigned*)mmPt;
  u16* dstPt = mmWS + (size_t)pt * KDIM;
#pragma unroll
  for (int ci = 0; ci < 4; ++ci) {
    const u16* apx = &gfbS[wave][ci * 16 + r16][0] + 8 * h;   // k = 8h..8h+7 | +16
    v16bf a = load16bf(apx, apx + 16);
    const u16* bpx = &wmatS[wave][r16][0] + 8 * h;            // row n=r16, k = 8h..8h+7 | +16
    v16bf b0 = load16bf(bpx, bpx + 16);
    v16bf b1 = load16bf(bpx + 16 * CMID, bpx + 16 * CMID + 16);
    v8f cz = {0.f, 0.f, 0.f, 0.f, 0.f, 0.f, 0.f, 0.f};
    v8f d0 = __builtin_amdgcn_wmma_f32_16x16x32_bf16(
        false, a, false, b0, (short)0, cz, false, false);
    v8f d1 = __builtin_amdgcn_wmma_f32_16x16x32_bf16(
        false, a, false, b1, (short)0, cz, false, false);
#pragma unroll
    for (int r = 0; r < 8; ++r) {
      int row = ci * 16 + r + 8 * h;       // C/D layout: VGPR r -> row r + 8h
      mm32[row * 16 + r16] = pack_bf16(d0[r], d1[r]);
    }
  }
  // Coalesced copy of this wave's 4 KB mm tile to global (8 x b128 per lane).
#pragma unroll
  for (int r = 0; r < 8; ++r) {
    int unit = r * 32 + lane;              // 16-byte units, contiguous per wave
    *(uint4*)(dstPt + unit * 8) = *(const uint4*)(mmPt + unit * 8);
  }
}

// ---------------------------------------------------------------------------
// Stage 2: out[16384 x 128] = mm[16384 x 2048] @ Wb[2048 x 128] + bias/BN.
// 256 threads = 8 waves = 4 strips x 2 column-halves; K in 32-chunks with a
// double-buffered async B stage (one barrier per chunk).
// ---------------------------------------------------------------------------
__global__ void __launch_bounds__(256) stage2_kernel(
    const u16* __restrict__ mmWS, const u16* __restrict__ Wb,
    const float* __restrict__ out_b, const float* __restrict__ out_g,
    const float* __restrict__ out_beta, const float* __restrict__ out_m,
    const float* __restrict__ out_v, float* __restrict__ out1) {
  __shared__ __attribute__((aligned(16))) u16 Bs[2][NS * COUT];  // 2 x 8 KB

  const int tid  = threadIdx.x;
  const int lane = tid & 31;
  const int wave = tid >> 5;                      // 0..7
  const int strip = blockIdx.x * 4 + (wave >> 1); // 16-point strip
  const int half = wave & 1;                      // column half: n-tiles 4h..4h+3
  const int pt0 = strip * 16;
  const int h = lane >> 4, rA = lane & 15;

  v8f acc[4];
#pragma unroll
  for (int i = 0; i < 4; ++i) {
    v8f z = {0.f, 0.f, 0.f, 0.f, 0.f, 0.f, 0.f, 0.f};
    acc[i] = z;
  }

  copy_chunk(Wb, &Bs[0][0], tid);                 // prologue: chunk 0 -> buf 0

  const u16* Abase = mmWS + (size_t)pt0 * KDIM;
  for (int kc = 0; kc < KDIM / NS; ++kc) {        // 64 chunks
    const int cur = kc & 1;
#if HAVE_ASYNC_LDS
    wait_async0();                                // own chunk-kc loads landed
#endif
    __syncthreads();                              // all landed; buf cur^1 free
    if (kc + 1 < KDIM / NS)
      copy_chunk(Wb + (kc + 1) * (NS * COUT), &Bs[cur ^ 1][0], tid);

    const u16* ap = Abase + rA * KDIM + kc * NS + 8 * h;   // A-frag (row-major)
    v16bf a = load16bf(ap, ap + 16);
#pragma unroll
    for (int nj = 0; nj < 4; ++nj) {
      const int ni = half * 4 + nj;
      const u16* bp = &Bs[cur][0] + (ni * 16 + rA) * NS + 8 * h;  // B-frag: col n, k = 8h..8h+7 | +16
      v16bf bfrag = load16bf(bp, bp + 16);
      acc[nj] = __builtin_amdgcn_wmma_f32_16x16x32_bf16(
          false, a, false, bfrag, (short)0, acc[nj], false, false);
    }
  }

  // ---- epilogue: +bias' (BN scale already folded into Wb), store float4 ----
  const int ocol = lane & 15;
  const int ptrow0 = pt0 + 8 * h;                 // rows r..r+7 are consecutive p
  const int bidx = ptrow0 >> 10;
  const int pbase = ptrow0 & (NP - 1);
#pragma unroll
  for (int nj = 0; nj < 4; ++nj) {
    int o = (half * 4 + nj) * 16 + ocol;
    float s = out_g[o] * rsqrtf(out_v[o] + 1e-5f);
    float bias = (out_b[o] - out_m[o]) * s + out_beta[o];
    float* dst = out1 + ((size_t)(bidx * COUT + o)) * NP + pbase;
    float4 v0 = make_float4(acc[nj][0] + bias, acc[nj][1] + bias,
                            acc[nj][2] + bias, acc[nj][3] + bias);
    float4 v1 = make_float4(acc[nj][4] + bias, acc[nj][5] + bias,
                            acc[nj][6] + bias, acc[nj][7] + bias);
    *(float4*)dst = v0;
    *(float4*)(dst + 4) = v1;
  }
}

// ---------------------------------------------------------------------------
extern "C" void kernel_launch(void* const* d_in, const int* in_sizes, int n_in,
                              void* d_out, int out_size, void* d_ws, size_t ws_size,
                              hipStream_t stream) {
  (void)in_sizes; (void)n_in; (void)out_size; (void)ws_size;

  const float* new_xyz  = (const float*)d_in[0];
  const float* gxyz     = (const float*)d_in[1];
  const float* gfeat    = (const float*)d_in[2];
  const float* wh_w     = (const float*)d_in[3];
  const float* wh_b     = (const float*)d_in[4];
  const float* wh_g     = (const float*)d_in[5];
  const float* wh_beta  = (const float*)d_in[6];
  const float* wh_m     = (const float*)d_in[7];
  const float* wh_v     = (const float*)d_in[8];
  const float* n1_w     = (const float*)d_in[9];
  const float* n1_b     = (const float*)d_in[10];
  const float* n1_g     = (const float*)d_in[11];
  const float* n1_beta  = (const float*)d_in[12];
  const float* n1_m     = (const float*)d_in[13];
  const float* n1_v     = (const float*)d_in[14];
  const float* n2_w     = (const float*)d_in[15];
  const float* n2_b     = (const float*)d_in[16];
  const float* n2_g     = (const float*)d_in[17];
  const float* n2_beta  = (const float*)d_in[18];
  const float* n2_m     = (const float*)d_in[19];
  const float* n2_v     = (const float*)d_in[20];
  const float* out_w    = (const float*)d_in[21];
  const float* out_b    = (const float*)d_in[22];
  const float* out_g    = (const float*)d_in[23];
  const float* out_beta = (const float*)d_in[24];
  const float* out_m    = (const float*)d_in[25];
  const float* out_v    = (const float*)d_in[26];

  u16* Wb   = (u16*)d_ws;                                              // 512 KB
  u16* mmWS = (u16*)((char*)d_ws + (size_t)KDIM * COUT * sizeof(u16)); // 64 MB

  float* out0 = (float*)d_out;                           // [B,NP,3]
  float* out1 = (float*)d_out + (size_t)BATCH * NP * 3;  // [B,COUT,NP]

  xyzT_kernel<<<(BATCH * NP * 3 + 255) / 256, 256, 0, stream>>>(new_xyz, out0);
  build_wb_kernel<<<(KDIM * COUT) / 256, 256, 0, stream>>>(out_w, out_g, out_v, Wb);
  stage1_kernel<<<PTS / 8, 256, 0, stream>>>(
      gxyz, gfeat, wh_w, wh_b, wh_g, wh_beta, wh_m, wh_v,
      n1_w, n1_b, n1_g, n1_beta, n1_m, n1_v,
      n2_w, n2_b, n2_g, n2_beta, n2_m, n2_v, mmWS);
  stage2_kernel<<<PTS / 16 / 4 / 4, 256, 0, stream>>>(   // 64 blocks x 8 waves
      mmWS, Wb, out_b, out_g, out_beta, out_m, out_v, out1);
}
